// FusionNet_17824114279027
// MI455X (gfx1250) — matrix operand order repaired and checked by reading, not compiled, not run on hardware
//
#include <hip/hip_runtime.h>
#include <hip/hip_bf16.h>
#include <math.h>

typedef __attribute__((ext_vector_type(16))) _Float16 v16h;
typedef __attribute__((ext_vector_type(8)))  _Float16 v8h;
typedef __attribute__((ext_vector_type(8)))  float    v8f;

#define EPSBN 1e-5f
#define NVOX (96*96*96)

__device__ __forceinline__ float sigmoidf_(float y) { return 1.0f / (1.0f + __expf(-y)); }

// ---------------------------------------------------------------------------
// Prep: mask + BN0-folded input features (C=1), as f16
// ---------------------------------------------------------------------------
__global__ void __launch_bounds__(256)
k_pre(const float* __restrict__ tsdf,
      const float* __restrict__ g0, const float* __restrict__ be0,
      const float* __restrict__ mu0, const float* __restrict__ var0,
      float* __restrict__ mask, _Float16* __restrict__ h0)
{
    int idx = blockIdx.x * 256 + threadIdx.x;
    if (idx >= NVOX) return;
    float sc = g0[0] * rsqrtf(var0[0] + EPSBN);
    float sh = be0[0] - mu0[0] * sc;
    float x = tsdf[idx];
    float m = (fabsf(x) < 1.0f) ? 1.0f : 0.0f;
    mask[idx] = m;
    h0[idx] = (_Float16)(m * (x * sc + sh));
}

// ---------------------------------------------------------------------------
// BN scale/shift tables for the fused epilogues (blocks 1,2,3)
// ---------------------------------------------------------------------------
__global__ void __launch_bounds__(64)
k_bnprep(const float* g1, const float* be1, const float* mu1, const float* var1,
         const float* g2, const float* be2, const float* mu2, const float* var2,
         const float* g3, const float* be3, const float* mu3, const float* var3,
         float* sc1, float* sh1, float* sc2, float* sh2, float* sc3, float* sh3)
{
    int t = threadIdx.x;
    if (t < 16) {
        float s = g1[t] * rsqrtf(var1[t] + EPSBN);
        sc1[t] = s; sh1[t] = be1[t] - mu1[t] * s;
    } else if (t < 48) {
        int c = t - 16;
        float s = g2[c] * rsqrtf(var2[c] + EPSBN);
        sc2[c] = s; sh2[c] = be2[c] - mu2[c] * s;
    } else {
        int c = t - 48;
        float s = g3[c] * rsqrtf(var3[c] + EPSBN);
        sc3[c] = s; sh3[c] = be3[c] - mu3[c] * s;
    }
}

// ---------------------------------------------------------------------------
// Pack layer-1 weights (32,16,3,3,3) into WMMA A-fragments.
// K = 2 taps * 16 cin (K=32). Lane L: m=L%16; element j:
// k = (L>=16?16:0) + j -> tap = 2p + (L>=16), cin = j. Tap 27 = zero pad.
// wp index: ((coblk*14 + p)*32 + lane)*16 + j
// ---------------------------------------------------------------------------
__global__ void __launch_bounds__(256)
k_pack1(const float* __restrict__ w, _Float16* __restrict__ wp)
{
    int t = blockIdx.x * 256 + threadIdx.x;
    if (t >= 2 * 14 * 32 * 16) return;
    int j    = t & 15;
    int lane = (t >> 4) & 31;
    int p    = (t >> 9) % 14;
    int cb   = t / 7168;
    int grp = j >> 3, wi = j & 7, hi = lane >> 4;
    int tap = 2 * p + hi;
    int cin = j;
    int co  = cb * 16 + (lane & 15);
    float v = (tap < 27) ? w[(co * 16 + cin) * 27 + tap] : 0.0f;
    wp[t] = (_Float16)v;
}

// ---------------------------------------------------------------------------
// Pack layer-2 weights (16,32,3,3,3): K = 32 cin per tap.
// wp index: (tap*32 + lane)*16 + j ; cin = (L>=16?16:0) + j
// ---------------------------------------------------------------------------
__global__ void __launch_bounds__(256)
k_pack2(const float* __restrict__ w, _Float16* __restrict__ wp)
{
    int t = blockIdx.x * 256 + threadIdx.x;
    if (t >= 27 * 32 * 16) return;
    int j    = t & 15;
    int lane = (t >> 4) & 31;
    int tap  = t >> 9;
    int grp = j >> 3, wi = j & 7, hi = lane >> 4;
    int cin = hi * 16 + j;
    int co  = lane & 15;
    wp[t] = (_Float16)w[(co * 32 + cin) * 27 + tap];
}

// ---------------------------------------------------------------------------
// Conv block 0 (1 -> 16), VALU (only 432 MAC/voxel). Epilogue fuses BN1.
// ---------------------------------------------------------------------------
__global__ void __launch_bounds__(256)
k_conv0(const _Float16* __restrict__ h0, const float* __restrict__ w,
        const float* __restrict__ bi, const float* __restrict__ sc,
        const float* __restrict__ sh, const float* __restrict__ mask,
        _Float16* __restrict__ h1)
{
    int idx = blockIdx.x * 256 + threadIdx.x;
    if (idx >= NVOX) return;
    int x = idx % 96, y = (idx / 96) % 96, z = idx / (96 * 96);
    float acc[16];
#pragma unroll
    for (int c = 0; c < 16; ++c) acc[c] = 0.0f;
#pragma unroll
    for (int tap = 0; tap < 27; ++tap) {
        int dz = tap / 9 - 1, dy = (tap % 9) / 3 - 1, dx = tap % 3 - 1;
        int zz = z + dz, yy = y + dy, xx = x + dx;
        if ((unsigned)zz < 96u && (unsigned)yy < 96u && (unsigned)xx < 96u) {
            float v = (float)h0[((zz * 96) + yy) * 96 + xx];
#pragma unroll
            for (int c = 0; c < 16; ++c) acc[c] += w[c * 27 + tap] * v;
        }
    }
    float m = mask[idx];
    v16h o;
#pragma unroll
    for (int c = 0; c < 16; ++c) {
        float s = sigmoidf_(acc[c] + bi[c]);
        o[c] = (_Float16)(m * (s * sc[c] + sh[c]));
    }
    *(v16h*)(h1 + (size_t)idx * 16) = o;
}

// ---------------------------------------------------------------------------
// Conv block 1 (16 -> 32), WMMA implicit GEMM with LDS-staged halo patch.
// Workgroup = 4 waves; patch = (d, h0..h0+3, w0..w0+15).
// Staged halo: 3(dz) x 6(h) x 18(x) voxels * 16ch f16 (zero-filled OOB),
// plus both Co-blocks of packed A fragments. Inner loop: pure ds_load + wmma.
// Tap addressing fully constant-folded: per iteration only a v_cndmask(hi)
// selects between the two compile-time tap offsets.
// ---------------------------------------------------------------------------
__global__ void __launch_bounds__(128)
k_conv1(const _Float16* __restrict__ hin, const _Float16* __restrict__ wp,
        const float* __restrict__ bi, const float* __restrict__ sc,
        const float* __restrict__ sh, const float* __restrict__ mask,
        _Float16* __restrict__ hout)
{
    __shared__ _Float16 sf[3 * 6 * 18 * 16];   // 5184 halves = 10368 B
    __shared__ _Float16 sw[2 * 14 * 32 * 16];  // 14336 halves = 28672 B

    const int tid = threadIdx.x;
    const int b   = blockIdx.x;
    const int seg = b % 6;
    const int hs  = (b / 6) % 24;
    const int d   = b / 144;
    const int h0  = hs * 4;
    const int w0  = seg * 16;

    // Stage feature halo: 648 chunks of 16B (2 chunks per voxel)
    for (int c = tid; c < 648; c += 128) {
        int vox = c >> 1, q = c & 1;
        int rx = vox % 18, t2 = vox / 18;
        int ry = t2 % 6,   rz = t2 / 6;
        int dd = d - 1 + rz, hh = h0 - 1 + ry, xx = w0 - 1 + rx;
        v8h v = {};
        if ((unsigned)dd < 96u && (unsigned)hh < 96u && (unsigned)xx < 96u)
            v = *(const v8h*)(hin + (size_t)(((dd * 96) + hh) * 96 + xx) * 16 + q * 8);
        *(v8h*)(sf + vox * 16 + q * 8) = v;
    }
    // Stage packed weights: 1792 chunks of 16B
    for (int c = tid; c < 1792; c += 128)
        *(v8h*)(sw + c * 8) = *(const v8h*)(wp + c * 8);
    __syncthreads();

    const int wid  = tid >> 5;       // 0..3 -> output row h0+wid
    const int lane = tid & 31;
    const int n  = lane & 15;
    const int hi = lane >> 4;

    // Lane-invariant part of the LDS B offset (halves): row(wid)*18*16 + n*16
    const int basewn = wid * (18 * 16) + n * 16;

    v8f acc0 = {};
    v8f acc1 = {};
#pragma unroll
    for (int p = 0; p < 14; ++p) {
        // Compile-time tap pair: lanes 0-15 use tap0=2p, lanes 16-31 tap1=2p+1.
        const int tap0 = 2 * p;
        const int tap1 = (2 * p + 1 > 26) ? 26 : (2 * p + 1);   // pad tap: A is zero
        const int c0 = (((tap0 / 9) * 6 + (tap0 % 9) / 3) * 18 + (tap0 % 3)) * 16;
        const int c1 = (((tap1 / 9) * 6 + (tap1 % 9) / 3) * 18 + (tap1 % 3)) * 16;
        const int boff = basewn + (hi ? c1 : c0);
        v16h bfrag = *(const v16h*)(sf + boff);
        v16h a0 = *(const v16h*)(sw + ((size_t)(0 * 14 + p) * 32 + lane) * 16);
        v16h a1 = *(const v16h*)(sw + ((size_t)(1 * 14 + p) * 32 + lane) * 16);
        acc0 = __builtin_amdgcn_wmma_f32_16x16x32_f16(false, a0, false, bfrag, (short)0, acc0, false, false);
        acc1 = __builtin_amdgcn_wmma_f32_16x16x32_f16(false, a1, false, bfrag, (short)0, acc1, false, false);
    }
    const int vox = ((d * 96) + h0 + wid) * 96 + w0 + n;
    const float m = mask[vox];
    const int cb = hi * 8;
    v8h o0, o1;
#pragma unroll
    for (int rr = 0; rr < 8; ++rr) {
        int co0 = cb + rr;
        float s0 = sigmoidf_(acc0[rr] + bi[co0]);
        o0[rr] = (_Float16)(m * (s0 * sc[co0] + sh[co0]));
        int co1 = 16 + cb + rr;
        float s1 = sigmoidf_(acc1[rr] + bi[co1]);
        o1[rr] = (_Float16)(m * (s1 * sc[co1] + sh[co1]));
    }
    *(v8h*)(hout + (size_t)vox * 32 + cb)      = o0;
    *(v8h*)(hout + (size_t)vox * 32 + 16 + cb) = o1;
}

// ---------------------------------------------------------------------------
// Conv block 2 (32 -> 16), WMMA with LDS-staged halo (features only: 20 KB;
// A fragments stay in global/L1).
// Tap offsets are pure unroll constants -> ds_load immediate offsets.
// ---------------------------------------------------------------------------
__global__ void __launch_bounds__(128)
k_conv2(const _Float16* __restrict__ hin, const _Float16* __restrict__ wp,
        const float* __restrict__ bi, const float* __restrict__ sc,
        const float* __restrict__ sh, const float* __restrict__ mask,
        _Float16* __restrict__ hout)
{
    __shared__ _Float16 sf[3 * 6 * 18 * 32];   // 10368 halves = 20736 B

    const int tid = threadIdx.x;
    const int b   = blockIdx.x;
    const int seg = b % 6;
    const int hs  = (b / 6) % 24;
    const int d   = b / 144;
    const int h0  = hs * 4;
    const int w0  = seg * 16;

    // Stage feature halo: 1296 chunks of 16B (4 chunks per voxel)
    for (int c = tid; c < 1296; c += 128) {
        int vox = c >> 2, q = c & 3;
        int rx = vox % 18, t2 = vox / 18;
        int ry = t2 % 6,   rz = t2 / 6;
        int dd = d - 1 + rz, hh = h0 - 1 + ry, xx = w0 - 1 + rx;
        v8h v = {};
        if ((unsigned)dd < 96u && (unsigned)hh < 96u && (unsigned)xx < 96u)
            v = *(const v8h*)(hin + (size_t)(((dd * 96) + hh) * 96 + xx) * 32 + q * 8);
        *(v8h*)(sf + vox * 32 + q * 8) = v;
    }
    __syncthreads();

    const int wid  = tid >> 5;
    const int lane = tid & 31;
    const int n  = lane & 15;
    const int hi = lane >> 4;

    // Lane-invariant part of the LDS B offset (halves)
    const int basewn = wid * (18 * 32) + n * 32 + hi * 16;

    v8f acc = {};
#pragma unroll
    for (int tap = 0; tap < 27; ++tap) {
        const int ctap = (((tap / 9) * 6 + (tap % 9) / 3) * 18 + (tap % 3)) * 32;
        v16h bfrag = *(const v16h*)(sf + basewn + ctap);
        v16h a = *(const v16h*)(wp + ((size_t)tap * 32 + lane) * 16);
        acc = __builtin_amdgcn_wmma_f32_16x16x32_f16(false, a, false, bfrag, (short)0, acc, false, false);
    }
    const int vox = ((d * 96) + h0 + wid) * 96 + w0 + n;
    const float m = mask[vox];
    const int cb = hi * 8;
    v8h o;
#pragma unroll
    for (int rr = 0; rr < 8; ++rr) {
        int co = cb + rr;
        float s = sigmoidf_(acc[rr] + bi[co]);
        o[rr] = (_Float16)(m * (s * sc[co] + sh[co]));
    }
    *(v8h*)(hout + (size_t)vox * 16 + cb) = o;
}

// ---------------------------------------------------------------------------
// Conv block 3 (16 -> 1), VALU. Final output: 2*mask*sigmoid(y) - 1 (f32).
// ---------------------------------------------------------------------------
__global__ void __launch_bounds__(256)
k_conv3(const _Float16* __restrict__ h3, const float* __restrict__ w,
        const float* __restrict__ bi, const float* __restrict__ mask,
        float* __restrict__ out)
{
    int idx = blockIdx.x * 256 + threadIdx.x;
    if (idx >= NVOX) return;
    int x = idx % 96, y = (idx / 96) % 96, z = idx / (96 * 96);
    float acc = 0.0f;
#pragma unroll
    for (int tap = 0; tap < 27; ++tap) {
        int dz = tap / 9 - 1, dy = (tap % 9) / 3 - 1, dx = tap % 3 - 1;
        int zz = z + dz, yy = y + dy, xx = x + dx;
        if ((unsigned)zz < 96u && (unsigned)yy < 96u && (unsigned)xx < 96u) {
            v16h hv = *(const v16h*)(h3 + (size_t)(((zz * 96) + yy) * 96 + xx) * 16);
#pragma unroll
            for (int c = 0; c < 16; ++c) acc += (float)hv[c] * w[c * 27 + tap];
        }
    }
    float s = sigmoidf_(acc + bi[0]);
    out[idx] = 2.0f * mask[idx] * s - 1.0f;
}

// ---------------------------------------------------------------------------
extern "C" void kernel_launch(void* const* d_in, const int* in_sizes, int n_in,
                              void* d_out, int out_size, void* d_ws, size_t ws_size,
                              hipStream_t stream) {
    (void)in_sizes; (void)n_in; (void)out_size; (void)ws_size;

    const float* tsdf = (const float*)d_in[0];
    // per-block params at 1 + 6*i : g, be, mu, var, w, bi
    const float* g0 = (const float*)d_in[1];  const float* be0 = (const float*)d_in[2];
    const float* mu0 = (const float*)d_in[3]; const float* var0 = (const float*)d_in[4];
    const float* w0 = (const float*)d_in[5];  const float* bi0 = (const float*)d_in[6];
    const float* g1 = (const float*)d_in[7];  const float* be1 = (const float*)d_in[8];
    const float* mu1 = (const float*)d_in[9]; const float* var1 = (const float*)d_in[10];
    const float* w1 = (const float*)d_in[11]; const float* bi1 = (const float*)d_in[12];
    const float* g2 = (const float*)d_in[13]; const float* be2 = (const float*)d_in[14];
    const float* mu2 = (const float*)d_in[15]; const float* var2 = (const float*)d_in[16];
    const float* w2 = (const float*)d_in[17]; const float* bi2 = (const float*)d_in[18];
    const float* g3 = (const float*)d_in[19]; const float* be3 = (const float*)d_in[20];
    const float* mu3 = (const float*)d_in[21]; const float* var3 = (const float*)d_in[22];
    const float* w3 = (const float*)d_in[23]; const float* bi3 = (const float*)d_in[24];
    float* out = (float*)d_out;

    // Carve workspace (all 256B aligned; total ~119 MB).
    char* ws = (char*)d_ws;
    size_t off = 0;
    auto carve = [&](size_t bytes) -> char* {
        char* p = ws + off;
        off = (off + bytes + 255) & ~(size_t)255;
        return p;
    };
    float*    mask = (float*)   carve((size_t)NVOX * 4);
    _Float16* h0   = (_Float16*)carve((size_t)NVOX * 2);
    _Float16* h1   = (_Float16*)carve((size_t)NVOX * 16 * 2);
    _Float16* h2   = (_Float16*)carve((size_t)NVOX * 32 * 2);
    _Float16* h3   = (_Float16*)carve((size_t)NVOX * 16 * 2);
    _Float16* wp1  = (_Float16*)carve((size_t)2 * 14 * 32 * 16 * 2);
    _Float16* wp2  = (_Float16*)carve((size_t)27 * 32 * 16 * 2);
    float* sc1 = (float*)carve(16 * 4); float* sh1 = (float*)carve(16 * 4);
    float* sc2 = (float*)carve(32 * 4); float* sh2 = (float*)carve(32 * 4);
    float* sc3 = (float*)carve(16 * 4); float* sh3 = (float*)carve(16 * 4);

    const int vb = NVOX / 256;          // 3456 blocks of 256
    const int tb = 96 * 24 * 6;         // 13824 blocks of 128 (4 waves, 1 patch each)

    k_bnprep<<<1, 64, 0, stream>>>(g1, be1, mu1, var1, g2, be2, mu2, var2,
                                   g3, be3, mu3, var3, sc1, sh1, sc2, sh2, sc3, sh3);
    k_pack1<<<(2 * 14 * 32 * 16 + 255) / 256, 256, 0, stream>>>(w1, wp1);
    k_pack2<<<(27 * 32 * 16 + 255) / 256, 256, 0, stream>>>(w2, wp2);
    k_pre<<<vb, 256, 0, stream>>>(tsdf, g0, be0, mu0, var0, mask, h0);
    k_conv0<<<vb, 256, 0, stream>>>(h0, w0, bi0, sc1, sh1, mask, h1);
    k_conv1<<<tb, 128, 0, stream>>>(h1, wp1, bi1, sc2, sh2, mask, h2);
    k_conv2<<<tb, 128, 0, stream>>>(h2, wp2, bi2, sc3, sh3, mask, h3);
    k_conv3<<<vb, 256, 0, stream>>>(h3, w3, bi3, mask, out);
}
